// GridSelfAttention_57363583205594
// MI455X (gfx1250) — compile-verified, matrix operand order checked by reading, not run on hardware
//
#include <hip/hip_runtime.h>
#include <hip/hip_bf16.h>

#define NRES  320
#define CCH   128
#define NN    (NRES * NRES)   // 102400
#define HEADS 4
#define DH    32

typedef __attribute__((ext_vector_type(16))) __bf16 v16bf;
typedef __attribute__((ext_vector_type(8)))  float  v8f;

union FragU { uint4 u[2]; v16bf v; };

// Load a 16x32 (A) or 32x16 (B) bf16 WMMA fragment from a row-major [rows][stride]
// bf16 array. Per ISA 7.12.2: lane L owns row (rowBase + L%16); its 16 halves are
// k = kBase + 8*(L/16) + [0..7]  (vgprs 0..3)  and  kBase + 16 + 8*(L/16) + [0..7]
// (vgprs 4..7): two contiguous 16-byte loads. A and B share this pattern.
__device__ __forceinline__ v16bf ld_frag(const __hip_bfloat16* __restrict__ p,
                                         int stride, int rowBase, int kBase) {
  const int lane = threadIdx.x & 31;
  const int row  = rowBase + (lane & 15);
  const int kk   = kBase + ((lane >> 4) << 3);
  FragU f;
  f.u[0] = *reinterpret_cast<const uint4*>(p + (size_t)row * stride + kk);
  f.u[1] = *reinterpret_cast<const uint4*>(p + (size_t)row * stride + kk + 16);
  return f.v;
}

__device__ __forceinline__ v8f wmma_bf16(v16bf a, v16bf b, v8f c) {
  return __builtin_amdgcn_wmma_f32_16x16x32_bf16(false, a, false, b,
                                                 (short)0, c, false, false);
}

// pack a C-fragment (8 f32, contiguous in memory after the M/N swap) into one
// 16-byte bf16 store
__device__ __forceinline__ void store_frag_packed(__hip_bfloat16* dst, const v8f& acc) {
  union { __hip_bfloat16 h[8]; uint4 u; } p;
#pragma unroll
  for (int i = 0; i < 8; ++i) p.h[i] = __float2bfloat16(acc[i]);
  *reinterpret_cast<uint4*>(dst) = p.u;
}

__device__ __forceinline__ float wave_sum(float v) {
#pragma unroll
  for (int m = 16; m; m >>= 1) v += __shfl_xor(v, m, 32);
  return v;
}
__device__ __forceinline__ float wave_max(float v) {
#pragma unroll
  for (int m = 16; m; m >>= 1) v = fmaxf(v, __shfl_xor(v, m, 32));
  return v;
}

// fast sigmoid: one v_exp + one v_rcp
__device__ __forceinline__ float fast_sigmoid(float x) {
  return __builtin_amdgcn_rcpf(1.0f + __expf(-x));
}

// ---------------------------------------------------------------------------
// Kernel 0: pack weights to bf16, transposed to [f][c] (feature-major) so they
// serve as A- or B-fragments with contiguous 16B loads. Fold 1/sqrt(D) into wq.
// ---------------------------------------------------------------------------
__global__ __launch_bounds__(256)
void prepack_kernel(const float* __restrict__ wq, const float* __restrict__ wk,
                    const float* __restrict__ wv, const float* __restrict__ wg,
                    const float* __restrict__ wo,
                    __hip_bfloat16* __restrict__ wqT, __hip_bfloat16* __restrict__ wkT,
                    __hip_bfloat16* __restrict__ wvT, __hip_bfloat16* __restrict__ wgT,
                    __hip_bfloat16* __restrict__ woT) {
  int idx = blockIdx.x * blockDim.x + threadIdx.x;
  if (idx >= CCH * CCH) return;
  int f = idx >> 7, c = idx & 127;            // f = out feature, c = in channel
  const float qscale = 0.17677669529663687f;  // 1/sqrt(32)
  wqT[f * CCH + c] = __float2bfloat16(wq[c * CCH + f] * qscale);
  wkT[f * CCH + c] = __float2bfloat16(wk[c * CCH + f]);
  wvT[f * CCH + c] = __float2bfloat16(wv[c * CCH + f]);
  wgT[f * CCH + c] = __float2bfloat16(wg[c * CCH + f]);
  woT[c * CCH + f] = __float2bfloat16(wo[f * CCH + c]);  // [out-ch][f]
}

// ---------------------------------------------------------------------------
// Kernel 1: fused LayerNorm + pair-bias + q/k/v/gate projections.
// Block = 4 waves, 64 flattened grid rows (16 per wave).
// Q/K/gate computed as Wt x Xt (A = weights, B = x) so C's M dim is the
// memory-contiguous feature dim -> one b128 store per fragment.
// V computed as X x Wv (M = grid row) because vT[d][n] is n-contiguous.
// ---------------------------------------------------------------------------
__global__ __launch_bounds__(128)
void ln_proj_kernel(const float* __restrict__ act,
                    const float* __restrict__ ln_scale, const float* __restrict__ ln_bias,
                    const float* __restrict__ wpb,       // [C][H]
                    const __hip_bfloat16* __restrict__ wqT, const __hip_bfloat16* __restrict__ wkT,
                    const __hip_bfloat16* __restrict__ wvT, const __hip_bfloat16* __restrict__ wgT,
                    const float* __restrict__ b_gate,
                    float* __restrict__ biasbuf,          // [H][N][N] f32
                    __hip_bfloat16* __restrict__ qbuf,    // [b][h][n][d]
                    __hip_bfloat16* __restrict__ kbuf,    // [b][h][n][d]
                    __hip_bfloat16* __restrict__ vT,      // [b][h][d][n]
                    __hip_bfloat16* __restrict__ gatebuf) // [row][f]
{
  __shared__ __hip_bfloat16 xn[64 * CCH];  // 16 KB
  const int wave = __builtin_amdgcn_readfirstlane(threadIdx.x >> 5);  // scalar
  const int lane = threadIdx.x & 31;
  const int half = lane >> 4;
  const int rowBase = blockIdx.x * 64;
  const int c0 = lane * 4;

  const float4 sc  = *(const float4*)(ln_scale + c0);
  const float4 bi  = *(const float4*)(ln_bias + c0);
  const float4 wb0 = *(const float4*)(wpb + (c0 + 0) * 4);
  const float4 wb1 = *(const float4*)(wpb + (c0 + 1) * 4);
  const float4 wb2 = *(const float4*)(wpb + (c0 + 2) * 4);
  const float4 wb3 = *(const float4*)(wpb + (c0 + 3) * 4);

  for (int r = 0; r < 16; ++r) {
    const int gi = rowBase + wave * 16 + r;
    const float4 x = *(const float4*)(act + (size_t)gi * CCH + c0);
    const float mu = wave_sum(x.x + x.y + x.z + x.w) * (1.0f / CCH);
    const float dx = x.x - mu, dy = x.y - mu, dz = x.z - mu, dw = x.w - mu;
    const float var = wave_sum(dx * dx + dy * dy + dz * dz + dw * dw) * (1.0f / CCH);
    const float rs = rsqrtf(var + 1e-5f);
    const float x0 = dx * rs * sc.x + bi.x;
    const float x1 = dy * rs * sc.y + bi.y;
    const float x2 = dz * rs * sc.z + bi.z;
    const float x3 = dw * rs * sc.w + bi.w;
    // pair bias: 4 dots over 128 channels
    const float p0 = wave_sum(x0 * wb0.x + x1 * wb1.x + x2 * wb2.x + x3 * wb3.x);
    const float p1 = wave_sum(x0 * wb0.y + x1 * wb1.y + x2 * wb2.y + x3 * wb3.y);
    const float p2 = wave_sum(x0 * wb0.z + x1 * wb1.z + x2 * wb2.z + x3 * wb3.z);
    const float p3 = wave_sum(x0 * wb0.w + x1 * wb1.w + x2 * wb2.w + x3 * wb3.w);
    if (lane == 0) {
      biasbuf[0 * NN + gi] = p0;
      biasbuf[1 * NN + gi] = p1;
      biasbuf[2 * NN + gi] = p2;
      biasbuf[3 * NN + gi] = p3;
    }
    __hip_bfloat16* xp = &xn[(wave * 16 + r) * CCH + c0];
    xp[0] = __float2bfloat16(x0); xp[1] = __float2bfloat16(x1);
    xp[2] = __float2bfloat16(x2); xp[3] = __float2bfloat16(x3);
  }
  __syncthreads();

  // cache the wave's 4 x-fragments (16 rows x full 128-channel contraction);
  // they serve as B (for Q/K/gate, swapped) or A (for V).
  v16bf xfr[4];
#pragma unroll
  for (int kc = 0; kc < 4; ++kc) xfr[kc] = ld_frag(xn, CCH, wave * 16, kc * 32);

  // 16 | 320: the wave's 16-row tile has one batch row bb; all scalar math.
  const int gi0 = rowBase + wave * 16;
  const int bb  = gi0 / NRES;          // uniform -> SALU
  const int n0  = gi0 - bb * NRES;
  const int r   = lane & 15;           // this lane's grid row (N index)

  // ---- Q projection (swapped): C[M=f][N=row]; q-scale folded into wqT ----
#pragma unroll
  for (int nt = 0; nt < 8; ++nt) {
    v8f acc = {};
#pragma unroll
    for (int kc = 0; kc < 4; ++kc)
      acc = wmma_bf16(ld_frag(wqT, CCH, nt * 16, kc * 32), xfr[kc], acc);
    const int h  = nt >> 1;                      // head owning this 16-feature tile
    const int d0 = (nt & 1) * 16 + 8 * half;     // contiguous d run
    store_frag_packed(qbuf + (((size_t)bb * HEADS + h) * NRES + n0 + r) * DH + d0, acc);
  }

  // ---- K projection (swapped) ----
#pragma unroll
  for (int nt = 0; nt < 8; ++nt) {
    v8f acc = {};
#pragma unroll
    for (int kc = 0; kc < 4; ++kc)
      acc = wmma_bf16(ld_frag(wkT, CCH, nt * 16, kc * 32), xfr[kc], acc);
    const int h  = nt >> 1;
    const int d0 = (nt & 1) * 16 + 8 * half;
    store_frag_packed(kbuf + (((size_t)bb * HEADS + h) * NRES + n0 + r) * DH + d0, acc);
  }

  // ---- V projection (original orientation): C[M=row][N=f]; vT[d][n] n-contig ----
#pragma unroll
  for (int nt = 0; nt < 8; ++nt) {
    v8f acc = {};
#pragma unroll
    for (int kc = 0; kc < 4; ++kc)
      acc = wmma_bf16(xfr[kc], ld_frag(wvT, CCH, nt * 16, kc * 32), acc);
    const int col = nt * 16 + r;
    const int h = col >> 5, d = col & 31;
    store_frag_packed(vT + (((size_t)bb * HEADS + h) * DH + d) * NRES + n0 + 8 * half, acc);
  }

  // ---- Gate projection + sigmoid (swapped): gatebuf[row][f] f-contig ----
#pragma unroll
  for (int nt = 0; nt < 8; ++nt) {
    v8f acc = {};
#pragma unroll
    for (int kc = 0; kc < 4; ++kc)
      acc = wmma_bf16(ld_frag(wgT, CCH, nt * 16, kc * 32), xfr[kc], acc);
    const int f0 = nt * 16 + 8 * half;
    const float4 bgA = *(const float4*)(b_gate + f0);
    const float4 bgB = *(const float4*)(b_gate + f0 + 4);
    v8f g;
    g[0] = fast_sigmoid(acc[0] + bgA.x); g[1] = fast_sigmoid(acc[1] + bgA.y);
    g[2] = fast_sigmoid(acc[2] + bgA.z); g[3] = fast_sigmoid(acc[3] + bgA.w);
    g[4] = fast_sigmoid(acc[4] + bgB.x); g[5] = fast_sigmoid(acc[5] + bgB.y);
    g[6] = fast_sigmoid(acc[6] + bgB.z); g[7] = fast_sigmoid(acc[7] + bgB.w);
    store_frag_packed(gatebuf + (size_t)(gi0 + r) * CCH + f0, g);
  }
}

// ---------------------------------------------------------------------------
// Kernel 2: attention + gating + fused output projection.
// Block = (b, 16-row q tile); one wave per head. All WMMAs in the swapped
// orientation so per-lane fragment I/O is contiguous (b128 / float4).
// LDS: 80K f32 logits + 40K bf16 probs + 4K bf16 wa  (< 320KB/WGP).
// ---------------------------------------------------------------------------
__global__ __launch_bounds__(128)
void attn_kernel(const __hip_bfloat16* __restrict__ qbuf, const __hip_bfloat16* __restrict__ kbuf,
                 const __hip_bfloat16* __restrict__ vT,   const __hip_bfloat16* __restrict__ gatebuf,
                 const float* __restrict__ biasbuf, const unsigned char* __restrict__ pmask,
                 const __hip_bfloat16* __restrict__ woT, const float* __restrict__ b_out,
                 float* __restrict__ out)
{
  __shared__ float          lg[HEADS * 16 * NRES];  // 81920 B
  __shared__ __hip_bfloat16 pb[HEADS * 16 * NRES];  // 40960 B
  __shared__ __hip_bfloat16 wa[16 * CCH];           //  4096 B

  const int qt   = blockIdx.x;        // 0..19
  const int b    = blockIdx.y;        // 0..319
  const int h    = __builtin_amdgcn_readfirstlane(threadIdx.x >> 5);  // wave == head
  const int lane = threadIdx.x & 31;
  const int half = lane >> 4;
  const int r    = lane & 15;         // this lane's q row within the tile

  const __hip_bfloat16* qh = qbuf + ((size_t)b * HEADS + h) * NRES * DH;
  const __hip_bfloat16* kh = kbuf + ((size_t)b * HEADS + h) * NRES * DH;
  const __hip_bfloat16* vh = vT   + ((size_t)b * HEADS + h) * DH * NRES;

  // hoist the mask column for this lane: depends only on k = lane + 32j and b
  bool maskv[10];
#pragma unroll
  for (int j = 0; j < 10; ++j)
    maskv[j] = pmask[(size_t)(lane + 32 * j) * NRES + b] != 0;

  // Q fragment (B operand; lane -> q row), covers full D=32 contraction
  const v16bf bq = ld_frag(qh, DH, qt * 16, 0);

  // QK^T swapped: C[M = k, N = q]; per-lane 8 contiguous k values
  for (int kt = 0; kt < 20; ++kt) {
    v16bf ak = ld_frag(kh, DH, kt * 16, 0);
    v8f c = {};
    c = wmma_bf16(ak, bq, c);
    const int k0 = kt * 16 + 8 * half;
    const float* bsrc = biasbuf + (size_t)h * NN + (qt * 16 + r) * NRES + k0;
    float* ldst = lg + (h * 16 + r) * NRES + k0;
    const float4 bA = *(const float4*)(bsrc);
    const float4 bB = *(const float4*)(bsrc + 4);
    float4 oA = { c[0] + bA.x, c[1] + bA.y, c[2] + bA.z, c[3] + bA.w };
    float4 oB = { c[4] + bB.x, c[5] + bB.y, c[6] + bB.z, c[7] + bB.w };
    *(float4*)(ldst)     = oA;
    *(float4*)(ldst + 4) = oB;
  }
  __syncthreads();

  // masked softmax over K=320, fp32
  for (int rr = 0; rr < 16; ++rr) {
    const float* lrow = lg + (h * 16 + rr) * NRES;
    float vals[10];
    float m = -3.4e38f;
#pragma unroll
    for (int j = 0; j < 10; ++j) {
      float v = lrow[lane + 32 * j];
      v = maskv[j] ? v : -1e9f;        // mask[b,...,k] = pair_mask[k][b]
      vals[j] = v;
      m = fmaxf(m, v);
    }
    m = wave_max(m);
    float s = 0.f;
#pragma unroll
    for (int j = 0; j < 10; ++j) { vals[j] = __expf(vals[j] - m); s += vals[j]; }
    s = wave_sum(s);
    const float inv = __builtin_amdgcn_rcpf(s);
#pragma unroll
    for (int j = 0; j < 10; ++j)
      pb[(h * 16 + rr) * NRES + lane + 32 * j] = __float2bfloat16(vals[j] * inv);
  }
  __syncthreads();

  // PV swapped: C[M = d, N = q];  A = vT rows (d), B = P rows (q)
  v8f acc0 = {}, acc1 = {};
  for (int kt = 0; kt < 10; ++kt) {
    v16bf ap  = ld_frag(pb + h * 16 * NRES, NRES, 0, kt * 32);
    v16bf av0 = ld_frag(vh, NRES, 0,  kt * 32);
    v16bf av1 = ld_frag(vh, NRES, 16, kt * 32);
    acc0 = wmma_bf16(av0, ap, acc0);
    acc1 = wmma_bf16(av1, ap, acc1);
  }

  // apply sigmoid gate; per-lane 8 contiguous f values -> b128 LDS store
  {
    const size_t grow = (size_t)b * NRES + qt * 16 + r;
#pragma unroll
    for (int t = 0; t < 2; ++t) {
      const int f0 = h * DH + t * 16 + 8 * half;
      const v8f acc = t ? acc1 : acc0;
      union { __hip_bfloat16 h8[8]; uint4 u; } gld;
      gld.u = *(const uint4*)(gatebuf + grow * CCH + f0);
      v8f gv;
#pragma unroll
      for (int i = 0; i < 8; ++i) gv[i] = acc[i] * __bfloat162float(gld.h8[i]);
      store_frag_packed(wa + r * CCH + f0, gv);
    }
  }
  __syncthreads();

  // fused output projection (swapped): C[M = out-ch, N = q]
  v16bf bwa[4];
#pragma unroll
  for (int kc = 0; kc < 4; ++kc) bwa[kc] = ld_frag(wa, CCH, 0, kc * 32);
#pragma unroll
  for (int t = 0; t < 2; ++t) {
    const int nt = h * 2 + t;
    v8f acc = {};
#pragma unroll
    for (int kc = 0; kc < 4; ++kc)
      acc = wmma_bf16(ld_frag(woT, CCH, nt * 16, kc * 32), bwa[kc], acc);
    const int co = nt * 16 + 8 * half;
    const float4 boA = *(const float4*)(b_out + co);
    const float4 boB = *(const float4*)(b_out + co + 4);
    float* odst = out + ((size_t)b * NRES + qt * 16 + r) * CCH + co;
    float4 oA = { acc[0] + boA.x, acc[1] + boA.y, acc[2] + boA.z, acc[3] + boA.w };
    float4 oB = { acc[4] + boB.x, acc[5] + boB.y, acc[6] + boB.z, acc[7] + boB.w };
    *(float4*)(odst)     = oA;
    *(float4*)(odst + 4) = oB;
  }
}

// ---------------------------------------------------------------------------
extern "C" void kernel_launch(void* const* d_in, const int* in_sizes, int n_in,
                              void* d_out, int out_size, void* d_ws, size_t ws_size,
                              hipStream_t stream) {
  (void)in_sizes; (void)n_in; (void)out_size; (void)ws_size;
  const float*         act  = (const float*)d_in[0];
  const unsigned char* pm   = (const unsigned char*)d_in[1];  // bool mask, 1B
  const float*         ln_s = (const float*)d_in[2];
  const float*         ln_b = (const float*)d_in[3];
  const float*         wpb  = (const float*)d_in[4];
  const float*         wq   = (const float*)d_in[5];
  const float*         wk   = (const float*)d_in[6];
  const float*         wv   = (const float*)d_in[7];
  const float*         wg   = (const float*)d_in[8];
  const float*         bg   = (const float*)d_in[9];
  const float*         wo   = (const float*)d_in[10];
  const float*         bo   = (const float*)d_in[11];
  float* out = (float*)d_out;

  // workspace carve-up (~102 MB total)
  char* ws = (char*)d_ws;
  size_t off = 0;
  auto take = [&](size_t bytes) {
    char* p = ws + off;
    off += (bytes + 255) & ~(size_t)255;
    return p;
  };
  __hip_bfloat16* wqT  = (__hip_bfloat16*)take((size_t)CCH * CCH * 2);
  __hip_bfloat16* wkT  = (__hip_bfloat16*)take((size_t)CCH * CCH * 2);
  __hip_bfloat16* wvT  = (__hip_bfloat16*)take((size_t)CCH * CCH * 2);
  __hip_bfloat16* wgT  = (__hip_bfloat16*)take((size_t)CCH * CCH * 2);
  __hip_bfloat16* woT  = (__hip_bfloat16*)take((size_t)CCH * CCH * 2);
  float*          bias = (float*)take((size_t)HEADS * NN * 4);
  __hip_bfloat16* qbuf = (__hip_bfloat16*)take((size_t)NRES * HEADS * NRES * DH * 2);
  __hip_bfloat16* kbuf = (__hip_bfloat16*)take((size_t)NRES * HEADS * NRES * DH * 2);
  __hip_bfloat16* vT   = (__hip_bfloat16*)take((size_t)NRES * HEADS * DH * NRES * 2);
  __hip_bfloat16* gate = (__hip_bfloat16*)take((size_t)NN * CCH * 2);

  prepack_kernel<<<(CCH * CCH + 255) / 256, 256, 0, stream>>>(
      wq, wk, wv, wg, wo, wqT, wkT, wvT, wgT, woT);

  ln_proj_kernel<<<NN / 64, 128, 0, stream>>>(
      act, ln_s, ln_b, wpb, wqT, wkT, wvT, wgT, bg,
      bias, qbuf, kbuf, vT, gate);

  attn_kernel<<<dim3(NRES / 16, NRES), 128, 0, stream>>>(
      qbuf, kbuf, vT, gate, bias, pm, woT, bo, out);
}
